// WindowedAttentionHead_30124900614801
// MI455X (gfx1250) — matrix operand order repaired and checked by reading, not compiled, not run on hardware
//
#include <hip/hip_runtime.h>

typedef __attribute__((ext_vector_type(16))) _Float16 v16h;
typedef __attribute__((ext_vector_type(8)))  _Float16 v8h;
typedef __attribute__((ext_vector_type(8)))  float    v8f;

#define WINSZ  64
#define NW     128          // windows per batch (8192/64)
#define SEQ    8192
#define CDIM   64
#define SHIFTN 32
#define NBATCH 64

__launch_bounds__(128, 1)
__global__ void swin_attn_kernel(const float* __restrict__ q,
                                 const float* __restrict__ k,
                                 const float* __restrict__ v,
                                 const float* __restrict__ bias_table,
                                 float* __restrict__ xout,
                                 float* __restrict__ attn_out)
{
    __shared__ __align__(32) _Float16 Qh[WINSZ * CDIM];     // Q window, row-major f16
    __shared__ __align__(32) _Float16 Kh[WINSZ * CDIM];     // K window, row-major f16
    __shared__ __align__(32) _Float16 Vt[CDIM * WINSZ];     // V window, TRANSPOSED (c-major) f16
    __shared__ __align__(32) _Float16 Pw[4][16 * WINSZ];    // per-wave softmax rows, f16
    __shared__ float biasS[2 * WINSZ];                      // 127 entries used

    const int tid  = threadIdx.x;
    const int lane = tid & 31;
    const int wv   = tid >> 5;           // wave id 0..3 -> tile-row of S/O
    const int blk  = blockIdx.x;
    const int b    = blk >> 7;           // batch index
    const int nwin = blk & (NW - 1);     // window index within batch

    const size_t batch_base = (size_t)b * SEQ * CDIM;
    const int    win_row0   = nwin * WINSZ;

    // ---- stage bias table ----
    if (tid < 2 * WINSZ - 1) biasS[tid] = bias_table[tid];

    // ---- stage Q,K (row-major) and V (transposed) as f16 in LDS ----
    {
        const int trow = tid >> 4;          // 0..7
        const int tcol = (tid & 15) * 4;    // 0..60
        #pragma unroll
        for (int rr = 0; rr < 8; ++rr) {
            const int row = rr * 8 + trow;                           // window-local row
            const int g   = (win_row0 + row + SHIFTN) & (SEQ - 1);   // cyclic shift
            const size_t gb = batch_base + (size_t)g * CDIM + tcol;
            const float4 fq = *(const float4*)(q + gb);
            const float4 fk = *(const float4*)(k + gb);
            const float4 fv = *(const float4*)(v + gb);
            _Float16* qd = &Qh[row * CDIM + tcol];
            qd[0] = (_Float16)fq.x; qd[1] = (_Float16)fq.y;
            qd[2] = (_Float16)fq.z; qd[3] = (_Float16)fq.w;
            _Float16* kd = &Kh[row * CDIM + tcol];
            kd[0] = (_Float16)fk.x; kd[1] = (_Float16)fk.y;
            kd[2] = (_Float16)fk.z; kd[3] = (_Float16)fk.w;
            Vt[(tcol + 0) * WINSZ + row] = (_Float16)fv.x;
            Vt[(tcol + 1) * WINSZ + row] = (_Float16)fv.y;
            Vt[(tcol + 2) * WINSZ + row] = (_Float16)fv.z;
            Vt[(tcol + 3) * WINSZ + row] = (_Float16)fv.w;
        }
    }
    __syncthreads();

    const int h = lane >> 4;     // lane-half selector
    const int n = lane & 15;     // tile column (B/C/D) or row (A)
    const int m = n;             // A-fragment row

    // ---- A fragments: Q rows 16*wv + m, two K-chunks of 32 ----
    // A k order used here (contiguous): halves 0..15 = cols kc*32 + 16h + (0..15)
    v16h aq[2];
    #pragma unroll
    for (int kc = 0; kc < 2; ++kc) {
        const _Float16* base = &Qh[(wv * 16 + m) * CDIM + kc * 32 + h * 16];
        const v8h lo = *(const v8h*)(base);
        const v8h hi = *(const v8h*)(base + 8);
        #pragma unroll
        for (int e = 0; e < 8; ++e) { aq[kc][e] = lo[e]; aq[kc][8 + e] = hi[e]; }
    }

    // ---- S = Q * K^T  (tile-row wv, 4 tile-cols, K-dim 64 = 2 x 32) ----
    // B k order used here (contiguous, as A): halves 0..15 = B[kc*32 + 16h + (0..15), n] = Kh[16*jt+n][...]
    v8f s[4];
    #pragma unroll
    for (int jt = 0; jt < 4; ++jt) {
        v8f acc = {};
        #pragma unroll
        for (int kc = 0; kc < 2; ++kc) {
            const v16h bk = *(const v16h*)(&Kh[(jt * 16 + n) * CDIM + kc * 32 + h * 16]);
            acc = __builtin_amdgcn_wmma_f32_16x16x32_f16(false, aq[kc], false, bk,
                                                         (short)0, acc, false, false);
        }
        s[jt] = acc;
    }

    // ---- scale + relative bias + shift mask ----
    const float SCALE = 0.125f;                // 64^-0.5
    const bool  do_mask = (nwin == NW - 1);    // only the wrap-around window is masked
    #pragma unroll
    for (int jt = 0; jt < 4; ++jt) {
        const int j = jt * 16 + n;
        #pragma unroll
        for (int r = 0; r < 8; ++r) {
            const int i = wv * 16 + h * 8 + r;         // window-local row of this element
            float val = s[jt][r] * SCALE + biasS[i - j + WINSZ - 1];
            if (do_mask && ((i < SHIFTN) != (j < SHIFTN))) val -= 100.0f;
            s[jt][r] = val;
        }
    }

    // ---- row-wise softmax (rows live in 16-lane groups; xor masks 1,2,4,8) ----
    float rinv[8];
    #pragma unroll
    for (int r = 0; r < 8; ++r) {
        float mx = fmaxf(fmaxf(s[0][r], s[1][r]), fmaxf(s[2][r], s[3][r]));
        #pragma unroll
        for (int msk = 1; msk < 16; msk <<= 1) mx = fmaxf(mx, __shfl_xor(mx, msk, 32));
        #pragma unroll
        for (int jt = 0; jt < 4; ++jt) s[jt][r] = __expf(s[jt][r] - mx);
        float sm = s[0][r] + s[1][r] + s[2][r] + s[3][r];
        #pragma unroll
        for (int msk = 1; msk < 16; msk <<= 1) sm += __shfl_xor(sm, msk, 32);
        rinv[r] = 1.0f / sm;
    }
    #pragma unroll
    for (int jt = 0; jt < 4; ++jt)
        #pragma unroll
        for (int r = 0; r < 8; ++r)
            s[jt][r] *= rinv[r];

    // ---- emit attn (f32, shifted frame) + stash P rows as f16 in LDS ----
    const size_t attn_base = (size_t)(b * NW + nwin) * (WINSZ * WINSZ);
    #pragma unroll
    for (int jt = 0; jt < 4; ++jt) {
        const int j = jt * 16 + n;
        #pragma unroll
        for (int r = 0; r < 8; ++r) {
            const int i = wv * 16 + h * 8 + r;
            attn_out[attn_base + (size_t)i * WINSZ + j] = s[jt][r];
            Pw[wv][(h * 8 + r) * WINSZ + j] = (_Float16)s[jt][r];
        }
    }
    __syncthreads();   // make P visible across lanes before re-fragmenting

    // ---- A fragments of P (this wave's own 16 rows) ----
    v16h ap[2];
    #pragma unroll
    for (int kc = 0; kc < 2; ++kc) {
        const _Float16* base = &Pw[wv][m * WINSZ + kc * 32 + h * 16];
        const v8h lo = *(const v8h*)(base);
        const v8h hi = *(const v8h*)(base + 8);
        #pragma unroll
        for (int e = 0; e < 8; ++e) { ap[kc][e] = lo[e]; ap[kc][8 + e] = hi[e]; }
    }

    // ---- O = P * V  (B fragments contiguous thanks to transposed V) ----
    #pragma unroll
    for (int ct = 0; ct < 4; ++ct) {
        v8f o = {};
        #pragma unroll
        for (int kc = 0; kc < 2; ++kc) {
            const v16h bv = *(const v16h*)(&Vt[(ct * 16 + n) * WINSZ + kc * 32 + h * 16]);
            o = __builtin_amdgcn_wmma_f32_16x16x32_f16(false, ap[kc], false, bv,
                                                       (short)0, o, false, false);
        }
        #pragma unroll
        for (int r = 0; r < 8; ++r) {
            const int i = wv * 16 + h * 8 + r;
            const int g = (win_row0 + i + SHIFTN) & (SEQ - 1);   // reverse cyclic shift
            xout[batch_base + (size_t)g * CDIM + ct * 16 + n] = o[r];
        }
    }
}

extern "C" void kernel_launch(void* const* d_in, const int* in_sizes, int n_in,
                              void* d_out, int out_size, void* d_ws, size_t ws_size,
                              hipStream_t stream) {
    const float* q  = (const float*)d_in[0];
    const float* k  = (const float*)d_in[1];
    const float* v  = (const float*)d_in[2];
    const float* bt = (const float*)d_in[3];

    float* xout = (float*)d_out;
    float* attn = (float*)d_out + (size_t)NBATCH * SEQ * CDIM;

    dim3 grid(NBATCH * NW);   // 8192 windows, one per 128-thread block
    dim3 block(128);
    swin_attn_kernel<<<grid, block, 0, stream>>>(q, k, v, bt, xout, attn);
}
